// VSSBlock_3702261809738
// MI455X (gfx1250) — matrix operand order repaired and checked by reading, not compiled, not run on hardware
//
#include <hip/hip_runtime.h>
#include <hip/hip_fp16.h>
#include <hip/hip_bf16.h>

typedef __attribute__((ext_vector_type(16))) _Float16 v16h;
typedef __attribute__((ext_vector_type(8)))  _Float16 v8h;
typedef __attribute__((ext_vector_type(8)))  float    v8f;

// ---------------- problem dims ----------------
#define B_   2
#define H_   64
#define W_   64
#define C_   192
#define DM_  384
#define N_   16
#define R_   12
#define K_   4
#define L_   4096            // H_*W_
#define CD_  44              // R_ + 2*N_
#define CH_  64              // scan chunk length
#define NCH_ 64              // L_/CH_

#define WMMA_F16(a, b, c) \
  __builtin_amdgcn_wmma_f32_16x16x32_f16(false, (a), false, (b), (short)0, (c), false, false)

// ---------------- WMMA fragment helpers (CDNA5 16x16x32 f16) ----------------
// A fragment (16x32 f16): lane m=lane&15, hi=lane>>4; reg r holds K pair
//   k = (r>>2)*16 + hi*8 + (r&3)*2  (ISA 7.12.2 16-bit A table)
// => per lane the fragment is two contiguous 16B chunks: halves [hi*8, hi*8+8)
//    and [16+hi*8, 16+hi*8+8). Row strides (192/384/768 f16) keep 16B alignment.
__device__ __forceinline__ v16h load_a_f16_fast(const _Float16* __restrict__ A, long ld) {
  int lane = threadIdx.x & 31;
  int m = lane & 15, hi = lane >> 4;
  const _Float16* p = A + (long)m * ld + hi * 8;
  v8h lo = *(const v8h*)p;
  v8h hh = *(const v8h*)(p + 16);
  v16h a;
#pragma unroll
  for (int i = 0; i < 8; ++i) { a[i] = lo[i]; a[i + 8] = hh[i]; }
  return a;
}

// gather variant from f32 with K tail (used only by the K=12 dt GEMM)
__device__ __forceinline__ v16h load_a_f32(const float* __restrict__ A, long ld, int kmax) {
  int lane = threadIdx.x & 31;
  int m = lane & 15, hi = lane >> 4;
  const float* p = A + (long)m * ld;
  v16h a;
#pragma unroll
  for (int r = 0; r < 8; ++r) {
    int k = ((r >> 2) << 4) + (hi << 3) + ((r & 3) << 1);
    a[2 * r]     = (k     < kmax) ? (_Float16)p[k]     : (_Float16)0.0f;
    a[2 * r + 1] = (k + 1 < kmax) ? (_Float16)p[k + 1] : (_Float16)0.0f;
  }
  return a;
}

// load one pre-packed B fragment tile (32 lanes x 16 halves, contiguous per lane)
__device__ __forceinline__ v16h load_packed(const _Float16* __restrict__ pk, int tile) {
  int lane = threadIdx.x & 31;
  return *(const v16h*)(pk + ((long)tile * 32 + lane) * 16);
}

// direction mapping (involution): source spatial index for scan position l,
// and also the spatial destination of scan output at position l.
__device__ __forceinline__ int dir_map(int k, int l) {
  int t = (k & 2) ? (L_ - 1 - l) : l;
  if (k & 1) t = ((t & 63) << 6) | (t >> 6);
  return t;
}

// ---------------- K0: pack an f32 weight matrix into B-fragment order (f16) --------
// B fragment layout: n = lane&15, K-half = lane>>4; reg r holds k = (r>>2)*16 + kh*8 + (r&3)*2, +1.
// dst tile index = (batch*nKB + kb)*nNT + nt ; per tile: 32 lanes x 16 halves.
__global__ void __launch_bounds__(128) pack_b(const float* __restrict__ src,
                                              _Float16* __restrict__ dst,
                                              int nBatch, int nKB, int nNT,
                                              long ldk, long ldn, long bStride,
                                              int Ktot, int Ntot) {
  int wid = blockIdx.x * 4 + (threadIdx.x >> 5);
  if (wid >= nBatch * nKB * nNT) return;
  int nt = wid % nNT;
  int t  = wid / nNT;
  int kb = t % nKB;
  int bt = t / nKB;
  int lane = threadIdx.x & 31;
  int n = lane & 15, kh = lane >> 4;
  const float* base = src + (long)bt * bStride + (long)(kb * 32) * ldk + (long)(nt * 16) * ldn;
  int kmax = Ktot - kb * 32;
  int nmax = Ntot - nt * 16;
  v16h b;
#pragma unroll
  for (int r = 0; r < 8; ++r) {
    int k = ((r >> 2) << 4) + (kh << 3) + ((r & 3) << 1);
    b[2 * r]     = (k     < kmax && n < nmax) ? (_Float16)base[(long)k * ldk + (long)n * ldn]
                                              : (_Float16)0.0f;
    b[2 * r + 1] = (k + 1 < kmax && n < nmax) ? (_Float16)base[(long)(k + 1) * ldk + (long)n * ldn]
                                              : (_Float16)0.0f;
  }
  *(v16h*)(dst + ((long)wid * 32 + lane) * 16) = b;
}

// ---------------- K1: pre-LN over C, emit f16 ----------------
__global__ void __launch_bounds__(256) ln_in_kernel(const float* __restrict__ in,
                                                    const float* __restrict__ g,
                                                    const float* __restrict__ bb,
                                                    _Float16* __restrict__ out) {
  int lane = threadIdx.x & 31;
  long row = (long)blockIdx.x * 8 + (threadIdx.x >> 5);
  if (row >= (long)B_ * L_) return;
  const float* p = in + row * C_;
  float v[6], s = 0.f, s2 = 0.f;
#pragma unroll
  for (int i = 0; i < 6; ++i) {
    float t = p[lane + 32 * i];
    v[i] = t; s += t; s2 += t * t;
  }
#pragma unroll
  for (int off = 16; off; off >>= 1) {
    s  += __shfl_xor(s, off, 32);
    s2 += __shfl_xor(s2, off, 32);
  }
  float mean = s / (float)C_;
  float inv  = rsqrtf(s2 / (float)C_ - mean * mean + 1e-6f);
  _Float16* o = out + row * C_;
#pragma unroll
  for (int i = 0; i < 6; ++i) {
    int c = lane + 32 * i;
    o[c] = (_Float16)((v[i] - mean) * inv * g[c] + bb[c]);
  }
}

// ---------------- K2: in_proj GEMM (8192x768x192) -> x, z; 4 N-tiles/wave ----------------
__global__ void __launch_bounds__(128) gemm_inproj(const _Float16* __restrict__ xn16,
                                                   const _Float16* __restrict__ pWin,
                                                   float* __restrict__ x, float* __restrict__ z) {
  const int TM = (B_ * L_) / 16;  // 512
  const int NG = 12;              // 48 N-tiles / 4
  int wid = blockIdx.x * 4 + (threadIdx.x >> 5);
  if (wid >= TM * NG) return;
  int mt = wid / NG, ng = wid % NG;
  int row0 = mt << 4;
  v8f acc[4] = {{}, {}, {}, {}};
  for (int kb = 0; kb < 6; ++kb) {
    v16h a = load_a_f16_fast(xn16 + (long)row0 * C_ + kb * 32, C_);
    int tb = kb * 48 + ng * 4;
#pragma unroll
    for (int j = 0; j < 4; ++j) {
      v16h b = load_packed(pWin, tb + j);
      acc[j] = WMMA_F16(a, b, acc[j]);
    }
  }
  int lane = threadIdx.x & 31;
  int n = lane & 15, hi = lane >> 4;
#pragma unroll
  for (int j = 0; j < 4; ++j) {
    int col = ng * 64 + j * 16 + n;
#pragma unroll
    for (int v = 0; v < 8; ++v) {
      long m = row0 + v + 8 * hi;
      if (col < DM_) x[m * DM_ + col] = acc[j][v];
      else           z[m * DM_ + (col - DM_)] = acc[j][v];
    }
  }
}

// ---------------- K3: depthwise 3x3 conv + SiLU, (B,L,DM)->(B,DM,L) ----------------
__global__ void __launch_bounds__(256) dwconv_silu(const float* __restrict__ x,
                                                   const float* __restrict__ w9,
                                                   const float* __restrict__ bias,
                                                   float* __restrict__ xc) {
  long idx = (long)blockIdx.x * 256 + threadIdx.x;
  if (idx >= (long)B_ * DM_ * L_) return;
  int l = (int)(idx % L_);
  long t = idx / L_;
  int d = (int)(t % DM_);
  int b = (int)(t / DM_);
  int h = l >> 6, w = l & 63;
  float acc = bias[d];
#pragma unroll
  for (int kh = 0; kh < 3; ++kh) {
    int hh = h + kh - 1;
    if ((unsigned)hh >= 64u) continue;
#pragma unroll
    for (int kw = 0; kw < 3; ++kw) {
      int ww = w + kw - 1;
      if ((unsigned)ww >= 64u) continue;
      acc += x[((long)b * L_ + hh * 64 + ww) * DM_ + d] * w9[d * 9 + kh * 3 + kw];
    }
  }
  xc[((long)b * DM_ + d) * L_ + l] = acc / (1.f + __expf(-acc));  // SiLU
}

// ---------------- K4: build direction-ordered activations, f16 (B,K,L,DM) ----------------
__global__ void __launch_bounds__(256) build_xsT(const float* __restrict__ xc,
                                                 _Float16* __restrict__ xsT) {
  long idx = (long)blockIdx.x * 256 + threadIdx.x;
  if (idx >= (long)B_ * K_ * L_ * DM_) return;
  int d = (int)(idx % DM_);
  long t = idx / DM_;
  int l = (int)(t % L_); t /= L_;
  int k = (int)(t % K_);
  int b = (int)(t / K_);
  xsT[idx] = (_Float16)xc[((long)b * DM_ + d) * L_ + dir_map(k, l)];
}

// ---------------- K5: x_proj GEMM per (b,k): (4096 x 44pad48 x 384); 3 N-tiles/wave ------
__global__ void __launch_bounds__(128) gemm_xproj(const _Float16* __restrict__ xsT,
                                                  const _Float16* __restrict__ pWx,
                                                  float* __restrict__ xdblT) {
  const int MT = L_ / 16;  // 256
  int wid = blockIdx.x * 4 + (threadIdx.x >> 5);
  if (wid >= B_ * K_ * MT) return;
  int bk = wid / MT;
  int mt = wid % MT;
  int k = bk % K_;
  int row0 = mt << 4;
  const _Float16* Abase = xsT + ((long)bk * L_ + row0) * DM_;
  v8f acc[3] = {{}, {}, {}};
  for (int kb = 0; kb < 12; ++kb) {
    v16h a = load_a_f16_fast(Abase + kb * 32, DM_);
    int tb = (k * 12 + kb) * 3;
#pragma unroll
    for (int j = 0; j < 3; ++j) {
      v16h b = load_packed(pWx, tb + j);
      acc[j] = WMMA_F16(a, b, acc[j]);
    }
  }
  int lane = threadIdx.x & 31;
  int n = lane & 15, hi = lane >> 4;
  float* Dp = xdblT + ((long)bk * L_ + row0) * CD_;
#pragma unroll
  for (int j = 0; j < 3; ++j) {
    int col = j * 16 + n;
    if (col < CD_) {
#pragma unroll
      for (int v = 0; v < 8; ++v) Dp[(long)(v + 8 * hi) * CD_ + col] = acc[j][v];
    }
  }
}

// ---------------- K6: dt GEMM (4096 x 384 x 12pad32) -> dtpre f16; 4 N-tiles/wave --------
__global__ void __launch_bounds__(128) gemm_dt(const float* __restrict__ xdblT,
                                               const _Float16* __restrict__ pWd,
                                               _Float16* __restrict__ dtpre) {
  const int MT = L_ / 16, NG = 6;  // 256, 24 N-tiles / 4
  int wid = blockIdx.x * 4 + (threadIdx.x >> 5);
  if (wid >= B_ * K_ * MT * NG) return;
  int ng = wid % NG;
  int t  = wid / NG;
  int mt = t % MT;
  int bk = t / MT;
  int k = bk % K_;
  int row0 = mt << 4;
  v16h a = load_a_f32(xdblT + ((long)bk * L_ + row0) * CD_, CD_, R_);
  v8f acc[4] = {{}, {}, {}, {}};
  int tb = k * 24 + ng * 4;
#pragma unroll
  for (int j = 0; j < 4; ++j) {
    v16h b = load_packed(pWd, tb + j);
    acc[j] = WMMA_F16(a, b, acc[j]);
  }
  int lane = threadIdx.x & 31;
  int n = lane & 15, hi = lane >> 4;
#pragma unroll
  for (int j = 0; j < 4; ++j) {
    int col = ng * 64 + j * 16 + n;
#pragma unroll
    for (int v = 0; v < 8; ++v)
      dtpre[((long)bk * L_ + row0 + v + 8 * hi) * DM_ + col] = (_Float16)acc[j][v];
  }
}

// ---------------- K7: scan pass 1 — per-chunk local scan (h | sum(delta)) ----------------
__global__ void __launch_bounds__(256) scan_pass1(const _Float16* __restrict__ dtpre,
                                                  const float* __restrict__ dtb,
                                                  const _Float16* __restrict__ xsT,
                                                  const float* __restrict__ xdblT,
                                                  const float* __restrict__ Alogs,
                                                  float* __restrict__ carryH,
                                                  float* __restrict__ carryS) {
  long g = (long)blockIdx.x * 256 + threadIdx.x;
  if (g >= (long)B_ * K_ * NCH_ * DM_) return;
  int d = (int)(g % DM_);
  long t = g / DM_;
  int c = (int)(t % NCH_); t /= NCH_;
  int k = (int)(t % K_);
  int b = (int)(t / K_);
  long bk = (long)b * K_ + k;
  float Ad[N_], h[N_];
#pragma unroll
  for (int n = 0; n < N_; ++n) {
    Ad[n] = -__expf(Alogs[((long)k * DM_ + d) * N_ + n]);
    h[n] = 0.f;
  }
  float bias = dtb[k * DM_ + d];
  float dsum = 0.f;
  for (int i = 0; i < CH_; ++i) {
    long row = bk * L_ + c * CH_ + i;
    float dp = (float)dtpre[row * DM_ + d] + bias;
    float delta = (dp > 20.f) ? dp : log1pf(__expf(dp));
    float du = delta * (float)xsT[row * DM_ + d];
    const float* Bv = xdblT + row * CD_ + R_;
    dsum += delta;
#pragma unroll
    for (int n = 0; n < N_; ++n)
      h[n] = __expf(delta * Ad[n]) * h[n] + du * Bv[n];
  }
  float* Hp = carryH + ((bk * NCH_ + c) * DM_ + d) * N_;
#pragma unroll
  for (int n = 0; n < N_; ++n) Hp[n] = h[n];
  carryS[(bk * NCH_ + c) * DM_ + d] = dsum;
}

// ---------------- K8: sequential cross-chunk combine (carryH -> chunk-initial h) --------
__global__ void __launch_bounds__(256) scan_combine(const float* __restrict__ Alogs,
                                                    float* __restrict__ carryH,
                                                    const float* __restrict__ carryS) {
  long g = (long)blockIdx.x * 256 + threadIdx.x;
  if (g >= (long)B_ * K_ * DM_) return;
  int d = (int)(g % DM_);
  long t = g / DM_;
  int k = (int)(t % K_);
  long bk = t;  // b*K_+k
  float Ad[N_], h[N_];
#pragma unroll
  for (int n = 0; n < N_; ++n) {
    Ad[n] = -__expf(Alogs[((long)k * DM_ + d) * N_ + n]);
    h[n] = 0.f;
  }
  for (int c = 0; c < NCH_; ++c) {
    float* Hp = carryH + ((bk * NCH_ + c) * DM_ + d) * N_;
    float s = carryS[(bk * NCH_ + c) * DM_ + d];
#pragma unroll
    for (int n = 0; n < N_; ++n) {
      float hl = Hp[n];          // chunk-local final state (h0 = 0)
      float hi = h[n];           // true initial state for this chunk
      Hp[n] = hi;                // overwrite with chunk-initial state
      h[n] = __expf(Ad[n] * s) * hi + hl;
    }
  }
}

// ---------------- K9: scan pass 2 — full scan with carry, y merged atomically ----------
__global__ void __launch_bounds__(256) scan_pass2(const _Float16* __restrict__ dtpre,
                                                  const float* __restrict__ dtb,
                                                  const _Float16* __restrict__ xsT,
                                                  const float* __restrict__ xdblT,
                                                  const float* __restrict__ Alogs,
                                                  const float* __restrict__ carryH,
                                                  float* __restrict__ ymerge) {
  long g = (long)blockIdx.x * 256 + threadIdx.x;
  if (g >= (long)B_ * K_ * NCH_ * DM_) return;
  int d = (int)(g % DM_);
  long t = g / DM_;
  int c = (int)(t % NCH_); t /= NCH_;
  int k = (int)(t % K_);
  int b = (int)(t / K_);
  long bk = (long)b * K_ + k;
  float Ad[N_], h[N_];
  const float* Hp = carryH + ((bk * NCH_ + c) * DM_ + d) * N_;
#pragma unroll
  for (int n = 0; n < N_; ++n) {
    Ad[n] = -__expf(Alogs[((long)k * DM_ + d) * N_ + n]);
    h[n] = Hp[n];
  }
  float bias = dtb[k * DM_ + d];
  for (int i = 0; i < CH_; ++i) {
    int l = c * CH_ + i;
    long row = bk * L_ + l;
    float dp = (float)dtpre[row * DM_ + d] + bias;
    float delta = (dp > 20.f) ? dp : log1pf(__expf(dp));
    float du = delta * (float)xsT[row * DM_ + d];
    const float* Bv = xdblT + row * CD_ + R_;
    const float* Cv = xdblT + row * CD_ + R_ + N_;
    float y = 0.f;
#pragma unroll
    for (int n = 0; n < N_; ++n) {
      h[n] = __expf(delta * Ad[n]) * h[n] + du * Bv[n];
      y += h[n] * Cv[n];
    }
    atomicAdd(&ymerge[((long)b * L_ + dir_map(k, l)) * DM_ + d], y);
  }
}

// ---------------- K10: + D-skip term, out-LN over DM, SiLU(z) gate -> f16 ----------------
__global__ void __launch_bounds__(256) merge_ln_gate(const float* __restrict__ ymerge,
                                                     const float* __restrict__ xc,
                                                     const float* __restrict__ Ds,
                                                     const float* __restrict__ g,
                                                     const float* __restrict__ bb,
                                                     const float* __restrict__ z,
                                                     _Float16* __restrict__ yg16) {
  int lane = threadIdx.x & 31;
  long row = (long)blockIdx.x * 8 + (threadIdx.x >> 5);
  if (row >= (long)B_ * L_) return;
  int b = (int)(row / L_);
  int l = (int)(row % L_);
  float v[12], s = 0.f, s2 = 0.f;
#pragma unroll
  for (int i = 0; i < 12; ++i) {
    int d = lane + 32 * i;
    float sd = Ds[d] + Ds[DM_ + d] + Ds[2 * DM_ + d] + Ds[3 * DM_ + d];
    float y = ymerge[row * DM_ + d] + sd * xc[((long)b * DM_ + d) * L_ + l];
    v[i] = y; s += y; s2 += y * y;
  }
#pragma unroll
  for (int off = 16; off; off >>= 1) {
    s  += __shfl_xor(s, off, 32);
    s2 += __shfl_xor(s2, off, 32);
  }
  float mean = s / (float)DM_;
  float inv  = rsqrtf(s2 / (float)DM_ - mean * mean + 1e-5f);
#pragma unroll
  for (int i = 0; i < 12; ++i) {
    int d = lane + 32 * i;
    float ln = (v[i] - mean) * inv * g[d] + bb[d];
    float zv = z[row * DM_ + d];
    yg16[row * DM_ + d] = (_Float16)(ln * (zv / (1.f + __expf(-zv))));
  }
}

// ---------------- K11: out_proj GEMM (8192x192x384) + residual; 4 N-tiles/wave ----------
__global__ void __launch_bounds__(128) gemm_out(const _Float16* __restrict__ yg16,
                                                const _Float16* __restrict__ pWo,
                                                const float* __restrict__ in,
                                                float* __restrict__ out) {
  const int TM = (B_ * L_) / 16;  // 512
  const int NG = 3;               // 12 N-tiles / 4
  int wid = blockIdx.x * 4 + (threadIdx.x >> 5);
  if (wid >= TM * NG) return;
  int mt = wid / NG, ng = wid % NG;
  int row0 = mt << 4;
  v8f acc[4] = {{}, {}, {}, {}};
  for (int kb = 0; kb < 12; ++kb) {
    v16h a = load_a_f16_fast(yg16 + (long)row0 * DM_ + kb * 32, DM_);
    int tb = kb * 12 + ng * 4;
#pragma unroll
    for (int j = 0; j < 4; ++j) {
      v16h b = load_packed(pWo, tb + j);
      acc[j] = WMMA_F16(a, b, acc[j]);
    }
  }
  int lane = threadIdx.x & 31;
  int n = lane & 15, hi = lane >> 4;
#pragma unroll
  for (int j = 0; j < 4; ++j) {
    int col = ng * 64 + j * 16 + n;
#pragma unroll
    for (int v = 0; v < 8; ++v) {
      long m = row0 + v + 8 * hi;
      out[m * C_ + col] = in[m * C_ + col] + acc[j][v];
    }
  }
}

// ---------------- launch ----------------
extern "C" void kernel_launch(void* const* d_in, const int* in_sizes, int n_in,
                              void* d_out, int out_size, void* d_ws, size_t ws_size,
                              hipStream_t stream) {
  const float* in      = (const float*)d_in[0];
  const float* norm_g  = (const float*)d_in[1];
  const float* norm_b  = (const float*)d_in[2];
  const float* Win     = (const float*)d_in[3];
  const float* conv_w  = (const float*)d_in[4];
  const float* conv_b  = (const float*)d_in[5];
  const float* Wx      = (const float*)d_in[6];
  const float* Wdt     = (const float*)d_in[7];
  const float* dt_b    = (const float*)d_in[8];
  const float* Alogs   = (const float*)d_in[9];
  const float* Ds      = (const float*)d_in[10];
  const float* on_g    = (const float*)d_in[11];
  const float* on_b    = (const float*)d_in[12];
  const float* Wo      = (const float*)d_in[13];
  float* out = (float*)d_out;

  char* p = (char*)d_ws;
  auto carve = [&](size_t bytes) -> void* {
    void* r = (void*)p;
    p += (bytes + 255) & ~(size_t)255;
    return r;
  };
  _Float16* xn16   = (_Float16*)carve((size_t)B_ * L_ * C_ * 2);
  float*    xbuf   = (float*)   carve((size_t)B_ * L_ * DM_ * 4);
  float*    zbuf   = (float*)   carve((size_t)B_ * L_ * DM_ * 4);
  float*    xconv  = (float*)   carve((size_t)B_ * DM_ * L_ * 4);
  _Float16* xsT16  = (_Float16*)carve((size_t)B_ * K_ * L_ * DM_ * 2);
  float*    xdblT  = (float*)   carve((size_t)B_ * K_ * L_ * CD_ * 4);
  _Float16* dtpre  = (_Float16*)carve((size_t)B_ * K_ * L_ * DM_ * 2);
  float*    carryH = (float*)   carve((size_t)B_ * K_ * NCH_ * DM_ * N_ * 4);
  float*    carryS = (float*)   carve((size_t)B_ * K_ * NCH_ * DM_ * 4);
  float*    ymerge = (float*)   carve((size_t)B_ * L_ * DM_ * 4);
  _Float16* yg16   = (_Float16*)carve((size_t)B_ * L_ * DM_ * 2);
  // packed weight fragments (tiles * 512 halves)
  _Float16* pWin = (_Float16*)carve((size_t)(1 * 6 * 48) * 512 * 2);   // 192x768
  _Float16* pWx  = (_Float16*)carve((size_t)(K_ * 12 * 3) * 512 * 2);  // 4x(384x44pad48)
  _Float16* pWd  = (_Float16*)carve((size_t)(K_ * 1 * 24) * 512 * 2);  // 4x(12pad32x384)
  _Float16* pWo  = (_Float16*)carve((size_t)(1 * 12 * 12) * 512 * 2);  // 384x192

  // 0) pack weights into WMMA B-fragment order (f16)
  pack_b<<<72, 128, 0, stream>>>(Win, pWin, 1, 6, 48, 2 * DM_, 1, 0, C_, 2 * DM_);
  pack_b<<<36, 128, 0, stream>>>(Wx,  pWx,  K_, 12, 3, 1, DM_, (long)CD_ * DM_, DM_, CD_);
  pack_b<<<24, 128, 0, stream>>>(Wdt, pWd,  K_, 1, 24, 1, R_, (long)DM_ * R_, R_, DM_);
  pack_b<<<36, 128, 0, stream>>>(Wo,  pWo,  1, 12, 12, C_, 1, 0, DM_, C_);

  // 1) pre-LN
  ln_in_kernel<<<(B_ * L_) / 8, 256, 0, stream>>>(in, norm_g, norm_b, xn16);
  // 2) in_proj GEMM (WMMA, 4 tiles/wave)
  gemm_inproj<<<(512 * 12) / 4, 128, 0, stream>>>(xn16, pWin, xbuf, zbuf);
  // 3) depthwise conv + SiLU
  dwconv_silu<<<((long)B_ * DM_ * L_) / 256, 256, 0, stream>>>(xbuf, conv_w, conv_b, xconv);
  // 4) cross-scan gather to f16
  build_xsT<<<((long)B_ * K_ * L_ * DM_) / 256, 256, 0, stream>>>(xconv, xsT16);
  // 5) x_proj GEMM (WMMA, 3 tiles/wave)
  gemm_xproj<<<(B_ * K_ * 256) / 4, 128, 0, stream>>>(xsT16, pWx, xdblT);
  // 6) dt GEMM (WMMA, 4 tiles/wave)
  gemm_dt<<<(B_ * K_ * 256 * 6) / 4, 128, 0, stream>>>(xdblT, pWd, dtpre);
  // 7) zero merge accumulator
  hipMemsetAsync(ymerge, 0, (size_t)B_ * L_ * DM_ * 4, stream);
  // 8-10) chunked parallel selective scan
  scan_pass1<<<((long)B_ * K_ * NCH_ * DM_) / 256, 256, 0, stream>>>(
      dtpre, dt_b, xsT16, xdblT, Alogs, carryH, carryS);
  scan_combine<<<((long)B_ * K_ * DM_) / 256, 256, 0, stream>>>(Alogs, carryH, carryS);
  scan_pass2<<<((long)B_ * K_ * NCH_ * DM_) / 256, 256, 0, stream>>>(
      dtpre, dt_b, xsT16, xdblT, Alogs, carryH, ymerge);
  // 11) merge + out-LN + gate
  merge_ln_gate<<<(B_ * L_) / 8, 256, 0, stream>>>(ymerge, xconv, Ds, on_g, on_b, zbuf, yg16);
  // 12) out_proj GEMM + residual (WMMA, 4 tiles/wave)
  gemm_out<<<(512 * 3) / 4, 128, 0, stream>>>(yg16, pWo, in, out);
}
